// Qwen3DecoderLayer_62431644614817
// MI455X (gfx1250) — matrix operand order repaired and checked by reading, not compiled, not run on hardware
//
#include <hip/hip_runtime.h>
#include <stdint.h>

// ---------------- problem constants (Qwen3-ish layer, B=1) ----------------
#define S_LEN 2048
#define H_DIM 4096
#define NQ_H  32
#define NKV_H 8
#define HD    128
#define I_DIM 12288
#define QKV_O 6144          // (NQ + 2*NKV) * D
#define EPSF  1e-6f
#define KSTEP 32

// ---------------- vector types ----------------
typedef __attribute__((ext_vector_type(16))) __bf16 v16bf;
typedef __attribute__((ext_vector_type(8)))  float  v8f;
typedef __attribute__((ext_vector_type(8)))  __bf16 bf16x8;
typedef __attribute__((ext_vector_type(16))) float  f32x16;

// ---------------- Tensor Data Mover availability ----------------
#if __has_builtin(__builtin_amdgcn_tensor_load_to_lds) && \
    __has_builtin(__builtin_amdgcn_s_wait_tensorcnt)
#define USE_TDM 1
#else
#define USE_TDM 0
#endif

#if USE_TDM
typedef __attribute__((ext_vector_type(4))) unsigned int u32x4;
typedef __attribute__((ext_vector_type(8))) int i32x8;
typedef __attribute__((ext_vector_type(4))) int i32x4;

// Issue a TDM 2D tile load: fp32 tile (tile_d0 x tile_d1) from a row-major
// tensor (dim0 contiguous = tensor_d0 elems, row stride = stride0 elems)
// into LDS at byte offset lds_off (rows packed contiguously).
// D# layout per CDNA5 ISA ch.8 (group0 128b, group1 256b).
__device__ __forceinline__ void tdm_load_tile_f32(
    unsigned lds_off, const float* gptr, int tensor_d0, int tensor_d1,
    int tile_d0, int tile_d1) {
  unsigned long long ga = (unsigned long long)(uintptr_t)gptr;
  u32x4 g0;
  g0[0] = 1u;                                   // count=1 (valid), user mode
  g0[1] = lds_off;                              // lds_addr [63:32]
  g0[2] = (unsigned)(ga & 0xFFFFFFFFu);         // global_addr [95:64]
  g0[3] = (unsigned)((ga >> 32) & 0x1FFFFFFu)   // global_addr [120:96]
          | (2u << 30);                         // type=2 ("image") [127:126]
  unsigned long long s0 = (unsigned long long)tensor_d0;  // dim0 stride
  i32x8 g1;
  g1[0] = (int)(2u << 16);                      // data_size=2 -> 4B elements
  g1[1] = (int)(((unsigned)tensor_d0 & 0xFFFFu) << 16);          // dim0 lo16
  g1[2] = (int)(((unsigned)tensor_d0 >> 16) |
                (((unsigned)tensor_d1 & 0xFFFFu) << 16));        // dim0 hi/dim1 lo
  g1[3] = (int)((((unsigned)tensor_d1 >> 16) & 0xFFFFu) |
                ((unsigned)tile_d0 << 16));                      // dim1 hi/tile0
  g1[4] = (int)(unsigned)tile_d1;               // tile_dim1 (tile_dim2=0)
  g1[5] = (int)(unsigned)(s0 & 0xFFFFFFFFu);    // dim0_stride [31:0]
  g1[6] = (int)(unsigned)((s0 >> 32) & 0xFFFFu);// dim0_stride [47:32]
  g1[7] = 0;                                    // dim1_stride hi (unused, 2D)
  i32x4 z4 = {0, 0, 0, 0};
#if __clang_major__ >= 23
  i32x8 z8 = {0, 0, 0, 0, 0, 0, 0, 0};
  __builtin_amdgcn_tensor_load_to_lds(g0, g1, z4, z4, z8, 0);
#else
  __builtin_amdgcn_tensor_load_to_lds(g0, g1, z4, z4, 0);
#endif
}
#endif  // USE_TDM

// native RNE float -> bf16 (lowers to v_cvt_pk_bf16_f32 on gfx1250)
__device__ __forceinline__ __bf16 f2bf(float f) { return (__bf16)f; }

__device__ __forceinline__ v8f v8f_zero() {
  v8f z;
#pragma unroll
  for (int i = 0; i < 8; ++i) z[i] = 0.f;
  return z;
}

__device__ __forceinline__ v8f wmma_bf16(v16bf a, v16bf b, v8f c) {
  // D(f32 16x16) = A(bf16 16x32) * B(bf16 32x16) + C
  return __builtin_amdgcn_wmma_f32_16x16x32_bf16(
      /*neg_a=*/false, a, /*neg_b=*/false, b,
      /*c_mod=*/(short)0, c, /*reuse_a=*/false, /*reuse_b=*/false);
}

// A-fragment (16x32 bf16): lane holds row M=lane%16, 16 contiguous K.
// lanes 0-15: K = k0..k0+15; lanes 16-31: K = k0+16..k0+31.
// Caller passes p = rowptr + k + (lane>=16 ? 16 : 0).
__device__ __forceinline__ v16bf load_a_bf16(const __bf16* p) {
  bf16x8 lo = *(const bf16x8*)p;
  bf16x8 hi = *(const bf16x8*)(p + 8);
  return __builtin_shufflevector(lo, hi, 0, 1, 2, 3, 4, 5, 6, 7, 8, 9, 10, 11,
                                 12, 13, 14, 15);
}

// B-fragment (32x16 bf16): lane holds column N=lane%16, 16 contiguous K.
// Caller passes p = colptr + k + (lane>=16 ? 16 : 0).
__device__ __forceinline__ v16bf load_b_bf16(const __bf16* p) {
  return *(const v16bf*)p;
}

// B-fragment from fp32 (global or LDS), packed cvt to bf16.
__device__ __forceinline__ v16bf load_b_f32(const float* p) {
  f32x16 v = *(const f32x16*)p;
  return __builtin_convertvector(v, v16bf);
}

// =====================================================================
// RMSNorm: fp32 [S,H] -> bf16 [S,H]
// =====================================================================
__global__ __launch_bounds__(256) void rmsnorm_kernel(
    const float* __restrict__ x, const float* __restrict__ w,
    __bf16* __restrict__ out, int H) {
  __shared__ float red[256];
  const int s = blockIdx.x;
  const int t = threadIdx.x;
  const float* xr = x + (size_t)s * H;
  float ss = 0.f;
  for (int i = t; i < H; i += 256) { float v = xr[i]; ss += v * v; }
  red[t] = ss;
  __syncthreads();
  for (int off = 128; off > 0; off >>= 1) {
    if (t < off) red[t] += red[t + off];
    __syncthreads();
  }
  const float rs = rsqrtf(red[0] / (float)H + EPSF);
  __bf16* orow = out + (size_t)s * H;
  for (int i = t; i < H; i += 256) orow[i] = f2bf(xr[i] * rs * w[i]);
}

// =====================================================================
// Generic WMMA GEMM: C[M,N] = A_bf16[M,K] * W_f32[N,K]^T
//   mode 0: Cf = acc                      (QKV, down)
//   mode 1: resid_out = acc + resid_in    (O-proj + residual)
//   mode 2: obf = bf16(silu(gbuf) * acc)  (up-proj fused with gate)
// Block: 256 thr = 8 waves; wave tile 64x32 (4x2 WMMA tiles); block 128x128.
// Grid: x = M-block, y = N-block (weight strips stay hot in 192MB L2).
// Weight tiles (128N x 32K fp32, 16KB) are staged into LDS by the Tensor
// Data Mover, double-buffered, tracked with TENSORcnt.
// =====================================================================
__global__ __launch_bounds__(256) void gemm_bf16_kernel(
    const __bf16* __restrict__ A, const float* __restrict__ W,
    int M, int N, int K, int mode,
    float* __restrict__ Cf, const float* __restrict__ resid_in,
    float* __restrict__ resid_out, const float* __restrict__ gbuf,
    __bf16* __restrict__ obf) {
#if USE_TDM
  __shared__ __align__(128) float ldsB[2][128 * KSTEP];  // 2 x 16KB
#endif
  const int lane = threadIdx.x & 31;
  const int wave = threadIdx.x >> 5;
  const int wm = wave >> 2;            // 0..1
  const int wn = wave & 3;             // 0..3
  const int lm = lane & 15;
  const int h8  = (lane >> 4) * 8;     // C row offset, also A prefetch K offset
  const int b16 = (lane >> 4) * 16;    // B-frag K offset
  const int mbase = blockIdx.x * 128 + wm * 64;
  const int nbase = blockIdx.y * 128 + wn * 32;

  v8f acc[4][2];
#pragma unroll
  for (int mi = 0; mi < 4; ++mi)
#pragma unroll
    for (int ni = 0; ni < 2; ++ni) acc[mi][ni] = v8f_zero();

  const __bf16* arow[4];
#pragma unroll
  for (int mi = 0; mi < 4; ++mi)
    arow[mi] = A + (size_t)(mbase + mi * 16 + lm) * K;

#if USE_TDM
  const unsigned ldsOff0 = (unsigned)(size_t)(void*)&ldsB[0][0];
  const unsigned ldsOff1 = (unsigned)(size_t)(void*)&ldsB[1][0];
  const float* wstrip = W + (size_t)(blockIdx.y * 128) * K;
  const int KT = K / KSTEP;
  if (wave == 0)
    tdm_load_tile_f32(ldsOff0, wstrip, K, 128, KSTEP, 128);
  for (int kt = 0; kt < KT; ++kt) {
    const int k = kt * KSTEP;
    if (wave == 0) {
      if (kt + 1 < KT) {
        tdm_load_tile_f32((kt & 1) ? ldsOff0 : ldsOff1,
                          wstrip + (size_t)(kt + 1) * KSTEP, K, 128, KSTEP,
                          128);
        __builtin_amdgcn_s_wait_tensorcnt(1);   // stage kt done, kt+1 in flight
      } else {
        __builtin_amdgcn_s_wait_tensorcnt(0);
      }
    }
    __syncthreads();                            // publish LDS stage kt
    const float* bbase = &ldsB[kt & 1][0];
    v16bf af[4], bfr[2];
#pragma unroll
    for (int mi = 0; mi < 4; ++mi) af[mi] = load_a_bf16(arow[mi] + k + b16);
#pragma unroll
    for (int ni = 0; ni < 2; ++ni)
      bfr[ni] = load_b_f32(bbase + (wn * 32 + ni * 16 + lm) * KSTEP + b16);
    if (kt + 1 < KT)
      __builtin_prefetch(arow[0] + k + KSTEP + h8, 0, 1);
#pragma unroll
    for (int mi = 0; mi < 4; ++mi)
#pragma unroll
      for (int ni = 0; ni < 2; ++ni)
        acc[mi][ni] = wmma_bf16(af[mi], bfr[ni], acc[mi][ni]);
    __syncthreads();                            // stage kt free for overwrite
  }
#else
  const float* wrow[2];
#pragma unroll
  for (int ni = 0; ni < 2; ++ni)
    wrow[ni] = W + (size_t)(nbase + ni * 16 + lm) * K;
  for (int k = 0; k < K; k += KSTEP) {
    v16bf af[4], bfr[2];
#pragma unroll
    for (int mi = 0; mi < 4; ++mi) af[mi] = load_a_bf16(arow[mi] + k + b16);
#pragma unroll
    for (int ni = 0; ni < 2; ++ni) bfr[ni] = load_b_f32(wrow[ni] + k + b16);
    if (k + KSTEP < K) {
      __builtin_prefetch(arow[0] + k + KSTEP + h8, 0, 1);
      __builtin_prefetch(wrow[0] + k + KSTEP + b16, 0, 1);
      __builtin_prefetch(wrow[1] + k + KSTEP + b16, 0, 1);
    }
#pragma unroll
    for (int mi = 0; mi < 4; ++mi)
#pragma unroll
      for (int ni = 0; ni < 2; ++ni)
        acc[mi][ni] = wmma_bf16(af[mi], bfr[ni], acc[mi][ni]);
  }
#endif

  // epilogue; C-tile element: row = r + 8*(lane>=16), col = lane%16
#pragma unroll
  for (int mi = 0; mi < 4; ++mi)
#pragma unroll
    for (int ni = 0; ni < 2; ++ni)
#pragma unroll
      for (int r = 0; r < 8; ++r) {
        const int row = mbase + mi * 16 + r + h8;
        const int col = nbase + ni * 16 + lm;
        const size_t idx = (size_t)row * N + col;
        const float v = acc[mi][ni][r];
        if (mode == 0) {
          Cf[idx] = v;
        } else if (mode == 1) {
          resid_out[idx] = v + resid_in[idx];
        } else {
          const float g = gbuf[idx];
          const float sg = g / (1.f + __expf(-g));
          obf[idx] = f2bf(sg * v);
        }
      }
}

// =====================================================================
// Per-head RMSNorm + RoPE on qkv fp32 [S, 6144]; emits:
//   qb  bf16 [S, NQ, D], kb bf16 [S, NKV, D], vT bf16 [NKV, D, S]
// grid = (48 heads, S), block = 128 (one lane per D element)
// =====================================================================
__global__ __launch_bounds__(128) void qkv_prep_kernel(
    const float* __restrict__ qkv, const int* __restrict__ pos,
    const float* __restrict__ qnw, const float* __restrict__ knw,
    __bf16* __restrict__ qb, __bf16* __restrict__ kb,
    __bf16* __restrict__ vT) {
  __shared__ float sq[128];
  __shared__ float buf[128];
  const int head = blockIdx.x;   // 0..47: [0,32) q, [32,40) k, [40,48) v
  const int s = blockIdx.y;
  const int t = threadIdx.x;
  const float v = qkv[(size_t)s * QKV_O + head * HD + t];

  if (head >= NQ_H + NKV_H) {    // V head: plain convert, transposed store
    const int kvh = head - (NQ_H + NKV_H);
    vT[((size_t)kvh * HD + t) * S_LEN + s] = f2bf(v);
    return;
  }
  sq[t] = v * v;
  __syncthreads();
  for (int off = 64; off > 0; off >>= 1) {
    if (t < off) sq[t] += sq[t + off];
    __syncthreads();
  }
  const float rs = rsqrtf(sq[0] * (1.f / (float)HD) + EPSF);
  const float* nw = (head < NQ_H) ? qnw : knw;
  buf[t] = v * rs * nw[t];
  __syncthreads();
  const int p = pos[s];
  const int j = t & 63;
  // inv_freq = THETA^(-j/64) = exp(-(j/64)*ln(1e6))
  const float inv = __expf(-((float)j * (1.f / 64.f)) * 13.815510557964274f);
  const float ang = (float)p * inv;
  const float c = __cosf(ang), sn = __sinf(ang);
  const float x1 = buf[j], x2 = buf[j + 64];
  const float o = (t < 64) ? (x1 * c - x2 * sn) : (x2 * c + x1 * sn);
  if (head < NQ_H)
    qb[(size_t)s * (NQ_H * HD) + head * HD + t] = f2bf(o);
  else
    kb[(size_t)s * (NKV_H * HD) + (head - NQ_H) * HD + t] = f2bf(o);
}

// =====================================================================
// Causal flash attention, bf16 WMMA, fp32 online softmax.
// grid = (S/16 q-tiles, NQ heads), block = 32 (one wave).
// Q kept in registers (4 A-frags); key tiles of 32; P transposed via LDS.
// =====================================================================
__global__ __launch_bounds__(32) void attn_kernel(
    const __bf16* __restrict__ qb, const __bf16* __restrict__ kb,
    const __bf16* __restrict__ vT, __bf16* __restrict__ out) {
  __shared__ __bf16 pshare[16 * 32];
  const int lane = threadIdx.x;
  const int qt = blockIdx.x;
  const int h  = blockIdx.y;
  const int kvh = h >> 2;             // NQ/NKV = 4
  const int lm  = lane & 15;
  const int hb  = lane >> 4;          // 0/1 half-wave
  const int h8  = hb * 8;
  const int b16 = hb * 16;
  const int q0  = qt * 16;
  const float scale = 0.08838834764831845f;   // D^-0.5

  // Q fragments: A-role, row = q0 + lane%16, contraction over d
  const __bf16* qrow = qb + (size_t)(q0 + lm) * (NQ_H * HD) + h * HD;
  v16bf qf[4];
#pragma unroll
  for (int dk = 0; dk < 4; ++dk) qf[dk] = load_a_bf16(qrow + dk * 32 + b16);

  v8f oacc[8];
#pragma unroll
  for (int d = 0; d < 8; ++d) oacc[d] = v8f_zero();
  float mrow[8], lrow[8];
#pragma unroll
  for (int r = 0; r < 8; ++r) { mrow[r] = -1e30f; lrow[r] = 0.f; }

  const int ntiles = (q0 + 15) / 32 + 1;       // causal: only needed key tiles
  for (int jt = 0; jt < ntiles; ++jt) {
    const int j0 = jt * 32;
    v8f sacc[2];
    sacc[0] = v8f_zero();
    sacc[1] = v8f_zero();
#pragma unroll
    for (int t = 0; t < 2; ++t) {
      // B-role: column = key j0 + t*16 + lane%16, contraction over d
      const __bf16* krow =
          kb + (size_t)(j0 + t * 16 + lm) * (NKV_H * HD) + kvh * HD;
#pragma unroll
      for (int dk = 0; dk < 4; ++dk) {
        v16bf kf = load_b_bf16(krow + dk * 32 + b16);
        sacc[t] = wmma_bf16(qf[dk], kf, sacc[t]);
      }
    }
    // online softmax; C rows: m = r + 8*hb, cols: lm and lm+16
#pragma unroll
    for (int r = 0; r < 8; ++r) {
      const int qg = q0 + r + h8;
      const float a = (j0 + lm      <= qg) ? sacc[0][r] * scale : -1e30f;
      const float b = (j0 + 16 + lm <= qg) ? sacc[1][r] * scale : -1e30f;
      float mx = fmaxf(a, b);
#pragma unroll
      for (int o = 1; o <= 8; o <<= 1) mx = fmaxf(mx, __shfl_xor(mx, o, 32));
      const float mn = fmaxf(mrow[r], mx);
      const float sc = __expf(mrow[r] - mn);
      const float p0 = __expf(a - mn);
      const float p1 = __expf(b - mn);
      float ps = p0 + p1;
#pragma unroll
      for (int o = 1; o <= 8; o <<= 1) ps += __shfl_xor(ps, o, 32);
      lrow[r] = lrow[r] * sc + ps;
      mrow[r] = mn;
#pragma unroll
      for (int d = 0; d < 8; ++d) oacc[d][r] *= sc;
      pshare[(r + h8) * 32 + lm]      = f2bf(p0);
      pshare[(r + h8) * 32 + 16 + lm] = f2bf(p1);
    }
    __syncthreads();
    // reload P in A-fragment layout (16 q-rows x 32 keys)
    v16bf pf = load_a_bf16(&pshare[lm * 32 + b16]);
    __syncthreads();
    // O += P * V ; V^T rows are contiguous in key index
#pragma unroll
    for (int dt = 0; dt < 8; ++dt) {
      const __bf16* vrow =
          vT + ((size_t)kvh * HD + dt * 16 + lm) * S_LEN + j0 + b16;
      v16bf vf = load_b_bf16(vrow);
      oacc[dt] = wmma_bf16(pf, vf, oacc[dt]);
    }
  }
  // normalize and store bf16 [S, NQ*D]
#pragma unroll
  for (int dt = 0; dt < 8; ++dt)
#pragma unroll
    for (int r = 0; r < 8; ++r) {
      const int qg = q0 + r + h8;
      out[(size_t)qg * (NQ_H * HD) + h * HD + dt * 16 + lm] =
          f2bf(oacc[dt][r] / lrow[r]);
    }
}

// =====================================================================
// Host-side orchestration
// =====================================================================
extern "C" void kernel_launch(void* const* d_in, const int* in_sizes, int n_in,
                              void* d_out, int out_size, void* d_ws,
                              size_t ws_size, hipStream_t stream) {
  const float* hidden  = (const float*)d_in[0];
  const int*   pos     = (const int*)d_in[1];
  const float* qkv_w   = (const float*)d_in[2];
  const float* o_w     = (const float*)d_in[3];
  const float* q_norm  = (const float*)d_in[4];
  const float* k_norm  = (const float*)d_in[5];
  const float* gate_w  = (const float*)d_in[6];
  const float* up_w    = (const float*)d_in[7];
  const float* down_w  = (const float*)d_in[8];
  const float* in_ln   = (const float*)d_in[9];
  const float* post_ln = (const float*)d_in[10];

  float* h_out = (float*)d_out;                    // output 0: [S,H]
  float* resid = h_out + (size_t)S_LEN * H_DIM;    // output 1: [S,H]

  char* p = (char*)d_ws;
  __bf16* hnorm = (__bf16*)p; p += (size_t)S_LEN * H_DIM * 2;
  float*  qkvb  = (float*)p;  p += (size_t)S_LEN * QKV_O * 4;
  __bf16* qbuf  = (__bf16*)p; p += (size_t)S_LEN * NQ_H * HD * 2;
  __bf16* kbuf  = (__bf16*)p; p += (size_t)S_LEN * NKV_H * HD * 2;
  __bf16* vTb   = (__bf16*)p; p += (size_t)NKV_H * HD * S_LEN * 2;
  __bf16* attnb = (__bf16*)p; p += (size_t)S_LEN * H_DIM * 2;
  __bf16* h2    = (__bf16*)p; p += (size_t)S_LEN * H_DIM * 2;
  float*  gbuf  = (float*)p;  p += (size_t)S_LEN * I_DIM * 4;
  __bf16* gubuf = (__bf16*)p; p += (size_t)S_LEN * I_DIM * 2;

  const dim3 blk256(256), blk128(128), blk32(32);

  // 1) input RMSNorm -> bf16
  rmsnorm_kernel<<<S_LEN, blk256, 0, stream>>>(hidden, in_ln, hnorm, H_DIM);
  // 2) QKV projection (grid.x = M-blocks for L2 weight reuse)
  gemm_bf16_kernel<<<dim3(S_LEN / 128, QKV_O / 128), blk256, 0, stream>>>(
      hnorm, qkv_w, S_LEN, QKV_O, H_DIM, 0, qkvb, nullptr, nullptr, nullptr,
      nullptr);
  // 3) per-head norm + RoPE + V transpose
  qkv_prep_kernel<<<dim3(NQ_H + 2 * NKV_H, S_LEN), blk128, 0, stream>>>(
      qkvb, pos, q_norm, k_norm, qbuf, kbuf, vTb);
  // 4) causal flash attention
  attn_kernel<<<dim3(S_LEN / 16, NQ_H), blk32, 0, stream>>>(qbuf, kbuf, vTb,
                                                            attnb);
  // 5) O projection + residual (writes output 1)
  gemm_bf16_kernel<<<dim3(S_LEN / 128, H_DIM / 128), blk256, 0, stream>>>(
      attnb, o_w, S_LEN, H_DIM, H_DIM, 1, nullptr, hidden, resid, nullptr,
      nullptr);
  // 6) post-attention RMSNorm
  rmsnorm_kernel<<<S_LEN, blk256, 0, stream>>>(resid, post_ln, h2, H_DIM);
  // 7) gate projection
  gemm_bf16_kernel<<<dim3(S_LEN / 128, I_DIM / 128), blk256, 0, stream>>>(
      h2, gate_w, S_LEN, I_DIM, H_DIM, 0, gbuf, nullptr, nullptr, nullptr,
      nullptr);
  // 8) up projection fused with silu(gate)*up -> bf16
  gemm_bf16_kernel<<<dim3(S_LEN / 128, I_DIM / 128), blk256, 0, stream>>>(
      h2, up_w, S_LEN, I_DIM, H_DIM, 2, nullptr, nullptr, nullptr, gbuf,
      gubuf);
  // 9) down projection (writes output 0)
  gemm_bf16_kernel<<<dim3(S_LEN / 128, H_DIM / 128), blk256, 0, stream>>>(
      gubuf, down_w, S_LEN, H_DIM, I_DIM, 0, h_out, nullptr, nullptr, nullptr,
      nullptr);

  (void)in_sizes; (void)n_in; (void)out_size; (void)ws_size;
}
